// Mamba2_76476187673114
// MI455X (gfx1250) — compile-verified, matrix operand order checked by reading, not run on hardware
//
#include <hip/hip_runtime.h>

// ---------------------------------------------------------------------------
// Mamba2 forward for MI455X (gfx1250, wave32, WMMA bf16 16x16x32, f32 accum)
// GEMMs fed by Tensor Data Mover (tensor_load_to_lds) double-buffered in LDS.
// ---------------------------------------------------------------------------

#define B_SZ    2
#define LSEQ    2048
#define DMODEL  2048
#define DSSM    4096
#define DSTATE  128
#define NHEADS  64
#define HEADDIM 64
#define NCHUNK  16
#define CHUNKL  128
#define DINP    8512          // 2*DSSM + 2*DSTATE + NHEADS
#define NPAD    8576          // DINP padded to multiple of 128
#define CONVD   4352          // DSSM + 2*DSTATE
#define BL      (B_SZ * LSEQ)

typedef __attribute__((ext_vector_type(16))) __bf16 v16bf;
typedef __attribute__((ext_vector_type(8)))  __bf16 v8bf;
typedef __attribute__((ext_vector_type(8)))  float  v8f;
typedef __attribute__((ext_vector_type(4)))  unsigned int u32x4;
typedef __attribute__((ext_vector_type(8)))  unsigned int u32x8;

// D = A(16x32) * B(32x16) + C  ; bf16 in, f32 out
__device__ __forceinline__ v8f wmma_bf16(v16bf a, v16bf b, v8f c) {
  return __builtin_amdgcn_wmma_f32_16x16x32_bf16(
      /*neg_a=*/false, a, /*neg_b=*/false, b,
      /*c_mod=*/(short)0, c, /*reuse_a=*/false, /*reuse_b=*/false);
}

// Load one 16x32 A-fragment (or B^T-fragment) for v_wmma_*_16x16x32.
// "base" is row-major [rows][ld] with contiguous K along a row.
__device__ __forceinline__ v16bf frag_ld(const __bf16* base, int ld, int row,
                                         int kbase, int lane) {
  const __bf16* p = base + (size_t)(row + (lane & 15)) * (size_t)ld
                         + (size_t)(kbase + ((lane >> 4) << 3));
  union { v16bf v; v8bf h[2]; } u;
  u.h[0] = *(const v8bf*)(p);
  u.h[1] = *(const v8bf*)(p + 16);
  return u.v;
}

// Store one 16x16 f32 C/D tile.
__device__ __forceinline__ void st_tile(float* C, int ldc, int r, int c, v8f v) {
#pragma unroll
  for (int j = 0; j < 8; ++j) C[(size_t)(r + j) * (size_t)ldc + c] = v[j];
}

__device__ __forceinline__ float siluf(float x) {
  return x / (1.f + __expf(-x));
}

// ---------------------------------------------------------------------------
// Tensor Data Mover: 2-D tile (rows x rowlen bf16 elems, row stride ld_elems)
// from global into LDS at byte offset lds_off. Wave-level op, TENSORcnt.
// D# group0: count=1 | lds_addr | global_addr | type=2
// D# group1: data_size=1(2B) | tensor dims | tile dims | dim0 stride
// ---------------------------------------------------------------------------
__device__ __forceinline__ void tdm_load_2d(const __bf16* gptr, unsigned lds_off,
                                            unsigned rows, unsigned rowlen,
                                            unsigned ld_elems) {
  unsigned long long ga = (unsigned long long)(uintptr_t)gptr;
  u32x4 g0;
  g0[0] = 1u;                                            // count=1, user D#
  g0[1] = lds_off;                                       // lds_addr (bytes)
  g0[2] = (unsigned)(ga & 0xffffffffu);                  // global_addr[31:0]
  g0[3] = (unsigned)((ga >> 32) & 0x01ffffffu) | (2u << 30); // addr[56:32]|type=2
  u32x8 g1;
  g1[0] = (1u << 16);                                    // data_size=1 (2 bytes)
  g1[1] = (rowlen & 0xffffu) << 16;                      // tensor_dim0 lo
  g1[2] = ((rowlen >> 16) & 0xffffu) | ((rows & 0xffffu) << 16); // td0 hi|td1 lo
  g1[3] = ((rows >> 16) & 0xffffu) | ((rowlen & 0xffffu) << 16); // td1 hi|tile_dim0
  g1[4] = rows & 0xffffu;                                // tile_dim1 (tile_dim2=0)
  g1[5] = ld_elems;                                      // tensor_dim0_stride lo
  g1[6] = 0u;                                            // stride0 hi | stride1 lo
  g1[7] = 0u;                                            // stride1 hi
  asm volatile("tensor_load_to_lds %0, %1" :: "s"(g0), "s"(g1) : "memory");
}

// ---------------------------------------------------------------------------
// casts
// ---------------------------------------------------------------------------
__global__ __launch_bounds__(256) void k_f32_to_bf16(const float* __restrict__ s,
                                                     __bf16* __restrict__ d, int n) {
  int i = blockIdx.x * 256 + threadIdx.x;
  if (i < n) d[i] = (__bf16)s[i];
}

__global__ __launch_bounds__(256) void k_cast_pad_rows(const float* __restrict__ s,
                                                       __bf16* __restrict__ d,
                                                       int rows_src, int cols, int rows_dst) {
  int i = blockIdx.x * 256 + threadIdx.x;
  if (i < rows_dst * cols) {
    int r = i / cols;
    d[i] = (r < rows_src) ? (__bf16)s[i] : (__bf16)0.f;
  }
}

// ---------------------------------------------------------------------------
// TDM-fed bf16 GEMM: C[M x N] = A[M x K] * B[N x K]^T  (f32 accumulate)
// block tile 64x128 (8 waves of 32x32); K-tile 64, LDS double buffered.
// Wave 0 drives the Tensor Data Mover; fragments come from LDS (ds_load_b128).
// ---------------------------------------------------------------------------
#define KT 64

__global__ __launch_bounds__(256) void k_gemm_tdm(const __bf16* __restrict__ A,
                                                  const __bf16* __restrict__ B,
                                                  float* __restrict__ C,
                                                  int K, int ldc) {
  __shared__ __bf16 As[2][64 * KT];    // 2 x 8 KB
  __shared__ __bf16 Bs[2][128 * KT];   // 2 x 16 KB

  const int lane = threadIdx.x & 31;
  const int wid  = threadIdx.x >> 5;
  const int wm   = wid >> 2;                 // 0..1
  const int wn   = wid & 3;                  // 0..3
  const int mblk = blockIdx.x * 64;
  const int nblk = blockIdx.y * 128;

  const __bf16* Ag = A + (size_t)mblk * (size_t)K;
  const __bf16* Bg = B + (size_t)nblk * (size_t)K;
  const unsigned As0 = (unsigned)(uintptr_t)(void*)&As[0][0];
  const unsigned As1 = (unsigned)(uintptr_t)(void*)&As[1][0];
  const unsigned Bs0 = (unsigned)(uintptr_t)(void*)&Bs[0][0];
  const unsigned Bs1 = (unsigned)(uintptr_t)(void*)&Bs[1][0];

  const int nk = K / KT;
  if (wid == 0) {                            // prologue: fill buffer 0
    tdm_load_2d(Ag, As0, 64, KT, (unsigned)K);
    tdm_load_2d(Bg, Bs0, 128, KT, (unsigned)K);
  }

  v8f a00 = {}, a01 = {}, a10 = {}, a11 = {};
  for (int kt = 0; kt < nk; ++kt) {
    if (wid == 0) __builtin_amdgcn_s_wait_tensorcnt(0);
    __syncthreads();                         // cur buffer ready; prev compute done
    if (wid == 0 && kt + 1 < nk) {           // prefetch next K-tile
      tdm_load_2d(Ag + (size_t)(kt + 1) * KT, (kt & 1) ? As0 : As1, 64, KT, (unsigned)K);
      tdm_load_2d(Bg + (size_t)(kt + 1) * KT, (kt & 1) ? Bs0 : Bs1, 128, KT, (unsigned)K);
    }
    const __bf16* At = As[kt & 1];
    const __bf16* Bt = Bs[kt & 1];
#pragma unroll
    for (int kc = 0; kc < KT; kc += 32) {
      v16bf fa0 = frag_ld(At, KT, wm * 32,      kc, lane);
      v16bf fa1 = frag_ld(At, KT, wm * 32 + 16, kc, lane);
      v16bf fb0 = frag_ld(Bt, KT, wn * 32,      kc, lane);
      v16bf fb1 = frag_ld(Bt, KT, wn * 32 + 16, kc, lane);
      a00 = wmma_bf16(fa0, fb0, a00);
      a01 = wmma_bf16(fa0, fb1, a01);
      a10 = wmma_bf16(fa1, fb0, a10);
      a11 = wmma_bf16(fa1, fb1, a11);
    }
  }

  const int r = mblk + wm * 32 + ((lane >> 4) << 3);
  const int c = nblk + wn * 32 + (lane & 15);
  st_tile(C, ldc, r,      c,      a00);
  st_tile(C, ldc, r,      c + 16, a01);
  st_tile(C, ldc, r + 16, c,      a10);
  st_tile(C, ldc, r + 16, c + 16, a11);
}

// ---------------------------------------------------------------------------
// depthwise causal conv1d (width 3) + SiLU on xBC slice of zxbcdt
// ---------------------------------------------------------------------------
__global__ __launch_bounds__(256) void k_conv(const float* __restrict__ zx,
                                              const float* __restrict__ cw,
                                              const float* __restrict__ cb,
                                              float* __restrict__ xc) {
  int idx = blockIdx.x * 256 + threadIdx.x;            // B*L*CONVD
  if (idx >= BL * CONVD) return;
  int c  = idx % CONVD;
  int bl = idx / CONVD;
  int l  = bl % LSEQ;
  float s = cb[c];
#pragma unroll
  for (int k = 0; k < 3; ++k) {
    int lp = l + k - 2;
    if (lp >= 0)
      s += zx[(size_t)(bl + (lp - l)) * NPAD + DSSM + c] * cw[c * 3 + k];
  }
  xc[(size_t)idx] = siluf(s);
}

// dt = softplus(raw + dt_bias); dA = dt * (-exp(A_log))
__global__ __launch_bounds__(256) void k_dt(const float* __restrict__ zx,
                                            const float* __restrict__ dt_bias,
                                            const float* __restrict__ A_log,
                                            float* __restrict__ dtv,
                                            float* __restrict__ dA) {
  int idx = blockIdx.x * 256 + threadIdx.x;            // B*L*NHEADS
  if (idx >= BL * NHEADS) return;
  int h  = idx & (NHEADS - 1);
  int bl = idx >> 6;
  float raw = zx[(size_t)bl * NPAD + (DINP - NHEADS) + h] + dt_bias[h];
  float d   = (raw > 20.f) ? raw : log1pf(__expf(raw));
  dtv[idx] = d;
  dA[idx]  = -d * __expf(A_log[h]);
}

// per-(b,h,chunk) inclusive cumsum of dA over the 128 chunk positions
__global__ __launch_bounds__(256) void k_acum(const float* __restrict__ dA,
                                              float* __restrict__ Acum,
                                              float* __restrict__ Atot) {
  int idx = blockIdx.x * 256 + threadIdx.x;            // B*NHEADS*NCHUNK
  if (idx >= B_SZ * NHEADS * NCHUNK) return;
  int c = idx & (NCHUNK - 1);
  int h = (idx >> 4) & (NHEADS - 1);
  int b = idx >> 10;
  float s = 0.f;
  for (int j = 0; j < CHUNKL; ++j) {
    s += dA[(size_t)((b * LSEQ + c * CHUNKL + j) * NHEADS) + h];
    Acum[(size_t)idx * CHUNKL + j] = s;
  }
  Atot[idx] = s;
}

// bf16 prep: Xdt[b,l,h,p] = x * dt ; Bbf / Cbf slices
__global__ __launch_bounds__(256) void k_prep(const float* __restrict__ xc,
                                              const float* __restrict__ dtv,
                                              __bf16* __restrict__ xdt,
                                              __bf16* __restrict__ Bbf,
                                              __bf16* __restrict__ Cbf) {
  int idx = blockIdx.x * 256 + threadIdx.x;            // B*L*CONVD
  if (idx >= BL * CONVD) return;
  int c  = idx % CONVD;
  int bl = idx / CONVD;
  float v = xc[(size_t)idx];
  if (c < DSSM) {
    int h = c >> 6;
    xdt[(size_t)bl * DSSM + c] = (__bf16)(v * dtv[(size_t)bl * NHEADS + h]);
  } else if (c < DSSM + DSTATE) {
    Bbf[(size_t)bl * DSTATE + (c - DSSM)] = (__bf16)v;
  } else {
    Cbf[(size_t)bl * DSTATE + (c - DSSM - DSTATE)] = (__bf16)v;
  }
}

// ---------------------------------------------------------------------------
// per-(b,c,h) chunk states: states[p,n] = sum_l Xdt[l,p]*decay[l]*Bm[l,n]
// ---------------------------------------------------------------------------
__global__ __launch_bounds__(256) void k_states(const __bf16* __restrict__ xdt,
                                                const __bf16* __restrict__ Bbf,
                                                const float* __restrict__ Acum,
                                                const float* __restrict__ Atot,
                                                float* __restrict__ states) {
  __shared__ __bf16 XwT[HEADDIM * CHUNKL];   // [p][l] 16 KB
  __shared__ __bf16 BmT[DSTATE * CHUNKL];    // [n][l] 32 KB
  __shared__ float  dec[CHUNKL];

  const int t    = threadIdx.x;
  const int bid  = blockIdx.x;               // B*NCHUNK*NHEADS
  const int h    = bid & (NHEADS - 1);
  const int c    = (bid >> 6) & (NCHUNK - 1);
  const int b    = bid >> 10;
  const size_t rowL = (size_t)(b * LSEQ + c * CHUNKL);
  const int    ahc  = ((b * NHEADS + h) * NCHUNK + c);

  if (t < CHUNKL)
    dec[t] = __expf(Atot[ahc] - Acum[(size_t)ahc * CHUNKL + t]);
  __syncthreads();

  for (int i = t; i < HEADDIM * CHUNKL; i += 256) {
    int p = i >> 7, l = i & 127;
    float x = (float)xdt[((rowL + l) * NHEADS + h) * HEADDIM + p];
    XwT[p * CHUNKL + l] = (__bf16)(x * dec[l]);
  }
  for (int i = t; i < DSTATE * CHUNKL; i += 256) {
    int n = i >> 7, l = i & 127;
    BmT[n * CHUNKL + l] = Bbf[(rowL + l) * DSTATE + n];
  }
  __syncthreads();

  const int lane = t & 31, wid = t >> 5;
  const int p0 = (wid & 3) * 16;             // 4 p-tiles
  const int n0 = (wid >> 2) * 64;            // 2 x 64 n-range
  v8f acc[4] = {};
  for (int k = 0; k < CHUNKL; k += 32) {
    v16bf fa = frag_ld(XwT, CHUNKL, p0, k, lane);
#pragma unroll
    for (int nt = 0; nt < 4; ++nt) {
      v16bf fb = frag_ld(BmT, CHUNKL, n0 + nt * 16, k, lane);
      acc[nt] = wmma_bf16(fa, fb, acc[nt]);
    }
  }
  float* S = states + (((size_t)(b * NCHUNK + c) * NHEADS + h) << 13); // *64*128
  const int pr = p0 + ((lane >> 4) << 3);
  const int nc0 = n0 + (lane & 15);
#pragma unroll
  for (int nt = 0; nt < 4; ++nt) st_tile(S, DSTATE, pr, nc0 + nt * 16, acc[nt]);
}

// inter-chunk scan: sin[z] = exp(Atot[z-1])*sin[z-1] + states[z-1], sin[0]=0
__global__ __launch_bounds__(256) void k_scan(const float* __restrict__ states,
                                              const float* __restrict__ Atot,
                                              __bf16* __restrict__ sinb) {
  int idx = blockIdx.x * 256 + threadIdx.x;  // B*NHEADS*HEADDIM*DSTATE
  if (idx >= B_SZ * NHEADS * HEADDIM * DSTATE) return;
  int n = idx & 127;
  int p = (idx >> 7) & 63;
  int h = (idx >> 13) & 63;
  int b = idx >> 19;
  const size_t cs   = (size_t)NHEADS * HEADDIM * DSTATE;  // chunk stride
  const size_t base = (size_t)b * NCHUNK * cs + ((size_t)h * HEADDIM + p) * DSTATE + n;
  float s = 0.f;
  for (int z = 0; z < NCHUNK; ++z) {
    sinb[base + (size_t)z * cs] = (__bf16)s;
    s = __expf(Atot[(b * NHEADS + h) * NCHUNK + z]) * s + states[base + (size_t)z * cs];
  }
}

// ---------------------------------------------------------------------------
// per-(b,c,h) SSD core: G = C*B^T, mask/decay, Y = G'*Xdt + exp(Acum)*(C*S^T) + D*x
// ---------------------------------------------------------------------------
__global__ __launch_bounds__(256) void k_ssd(const __bf16* __restrict__ Cbf,
                                             const __bf16* __restrict__ Bbf,
                                             const __bf16* __restrict__ xdt,
                                             const __bf16* __restrict__ sinb,
                                             const float* __restrict__ Acum,
                                             const float* __restrict__ xc,
                                             const float* __restrict__ Dv,
                                             float* __restrict__ Y) {
  __shared__ __bf16 Gsh[CHUNKL * CHUNKL];    // [l][s] 32 KB
  __shared__ __bf16 XdT[HEADDIM * CHUNKL];   // [p][s] 16 KB
  __shared__ float  Ach[CHUNKL];

  const int t    = threadIdx.x;
  const int lane = t & 31, wid = t >> 5;
  const int bid  = blockIdx.x;
  const int h    = bid & (NHEADS - 1);
  const int c    = (bid >> 6) & (NCHUNK - 1);
  const int b    = bid >> 10;
  const size_t rowL = (size_t)(b * LSEQ + c * CHUNKL);
  const int    ahc  = ((b * NHEADS + h) * NCHUNK + c);

  if (t < CHUNKL) Ach[t] = Acum[(size_t)ahc * CHUNKL + t];
  for (int i = t; i < HEADDIM * CHUNKL; i += 256) {
    int p = i >> 7, s = i & 127;
    XdT[p * CHUNKL + s] = xdt[((rowL + s) * NHEADS + h) * HEADDIM + p];
  }

  // G strip (rows l0..l0+15, all 128 s), straight from global bf16
  const __bf16* Cg = Cbf + rowL * DSTATE;
  const __bf16* Bg = Bbf + rowL * DSTATE;
  const int l0 = wid * 16;
  v8f g[8] = {};
  for (int k = 0; k < DSTATE; k += 32) {
    v16bf fa = frag_ld(Cg, DSTATE, l0, k, lane);
#pragma unroll
    for (int st = 0; st < 8; ++st) {
      v16bf fb = frag_ld(Bg, DSTATE, st * 16, k, lane);
      g[st] = wmma_bf16(fa, fb, g[st]);
    }
  }
  __syncthreads();                           // Ach / XdT ready

  // causal decay mask -> G' (bf16 in LDS)
  const int lr = l0 + ((lane >> 4) << 3);
  const int sc = lane & 15;
#pragma unroll
  for (int st = 0; st < 8; ++st) {
#pragma unroll
    for (int j = 0; j < 8; ++j) {
      int l = lr + j, s = st * 16 + sc;
      float v = (l >= s) ? g[st][j] * __expf(Ach[l] - Ach[s]) : 0.f;
      Gsh[l * CHUNKL + s] = (__bf16)v;
    }
  }
  __syncthreads();

  // Y_diag = G' * Xdt ; Y_off = C * states_in^T
  const __bf16* Sg = sinb + (((size_t)(b * NCHUNK + c) * NHEADS + h) << 13);
  v8f yd[4] = {}, yo[4] = {};
  for (int k = 0; k < CHUNKL; k += 32) {
    v16bf fag = frag_ld(Gsh, CHUNKL, l0, k, lane);
    v16bf fac = frag_ld(Cg, DSTATE, l0, k, lane);
#pragma unroll
    for (int pt = 0; pt < 4; ++pt) {
      v16bf fbx = frag_ld(XdT, CHUNKL, pt * 16, k, lane);
      yd[pt] = wmma_bf16(fag, fbx, yd[pt]);
      v16bf fbs = frag_ld(Sg, DSTATE, pt * 16, k, lane);
      yo[pt] = wmma_bf16(fac, fbs, yo[pt]);
    }
  }

  const float Dh = Dv[h];
#pragma unroll
  for (int pt = 0; pt < 4; ++pt) {
#pragma unroll
    for (int j = 0; j < 8; ++j) {
      int l = lr + j, p = pt * 16 + sc;
      float x = xc[(rowL + l) * CONVD + h * HEADDIM + p];
      Y[((rowL + l) * NHEADS + h) * HEADDIM + p] =
          yd[pt][j] + __expf(Ach[l]) * yo[pt][j] + Dh * x;
    }
  }
}

// gated RMSNorm: y *= silu(z); y *= rsqrt(mean(y^2)+eps)*norm_w ; -> bf16
__global__ __launch_bounds__(256) void k_gatenorm(const float* __restrict__ Y,
                                                  const float* __restrict__ zx,
                                                  const float* __restrict__ norm_w,
                                                  __bf16* __restrict__ yb) {
  __shared__ float red[256];
  const size_t row = blockIdx.x;             // B*L
  const int t = threadIdx.x;
  float vals[16];
  float ss = 0.f;
#pragma unroll
  for (int i = 0; i < 16; ++i) {
    int e = t + i * 256;
    float y = Y[row * DSSM + e];
    float z = zx[row * NPAD + e];
    y *= siluf(z);
    vals[i] = y;
    ss += y * y;
  }
  red[t] = ss;
  __syncthreads();
  for (int o = 128; o > 0; o >>= 1) {
    if (t < o) red[t] += red[t + o];
    __syncthreads();
  }
  const float scale = rsqrtf(red[0] / (float)DSSM + 1e-5f);
#pragma unroll
  for (int i = 0; i < 16; ++i) {
    int e = t + i * 256;
    yb[row * DSSM + e] = (__bf16)(vals[i] * scale * norm_w[e]);
  }
}

// out = tmp + tmp reversed along L
__global__ __launch_bounds__(256) void k_final(const float* __restrict__ tmp,
                                               float* __restrict__ out) {
  int idx = blockIdx.x * 256 + threadIdx.x;  // B*L*DMODEL
  if (idx >= BL * DMODEL) return;
  int d = idx % DMODEL;
  int l = (idx / DMODEL) % LSEQ;
  int b = idx / (DMODEL * LSEQ);
  out[(size_t)idx] = tmp[(size_t)idx] +
                     tmp[((size_t)b * LSEQ + (LSEQ - 1 - l)) * DMODEL + d];
}

// ---------------------------------------------------------------------------
// host launch
// ---------------------------------------------------------------------------
extern "C" void kernel_launch(void* const* d_in, const int* in_sizes, int n_in,
                              void* d_out, int out_size, void* d_ws, size_t ws_size,
                              hipStream_t stream) {
  (void)in_sizes; (void)n_in; (void)out_size; (void)ws_size;
  const float* u       = (const float*)d_in[0];
  const float* W_in    = (const float*)d_in[1];
  const float* conv_w  = (const float*)d_in[2];
  const float* conv_b  = (const float*)d_in[3];
  const float* dt_bias = (const float*)d_in[4];
  const float* A_log   = (const float*)d_in[5];
  const float* Dv      = (const float*)d_in[6];
  const float* norm_w  = (const float*)d_in[7];
  const float* W_out   = (const float*)d_in[8];
  float* out = (float*)d_out;

  char* ws = (char*)d_ws;
  size_t off = 0;
  auto alloc = [&](size_t bytes) -> char* {
    char* p = ws + off;
    off = (off + bytes + 255) & ~(size_t)255;
    return p;
  };
  __bf16* u_bf    = (__bf16*)alloc((size_t)BL * DMODEL * 2);
  __bf16* Win_bf  = (__bf16*)alloc((size_t)NPAD * DMODEL * 2);
  float*  zx      = (float*) alloc((size_t)BL * NPAD * 4);
  float*  xc      = (float*) alloc((size_t)BL * CONVD * 4);
  float*  dtv     = (float*) alloc((size_t)BL * NHEADS * 4);
  float*  dA      = (float*) alloc((size_t)BL * NHEADS * 4);
  float*  Acum    = (float*) alloc((size_t)B_SZ * NHEADS * NCHUNK * CHUNKL * 4);
  float*  Atot    = (float*) alloc((size_t)B_SZ * NHEADS * NCHUNK * 4);
  __bf16* xdt     = (__bf16*)alloc((size_t)BL * DSSM * 2);
  __bf16* Bbf     = (__bf16*)alloc((size_t)BL * DSTATE * 2);
  __bf16* Cbf     = (__bf16*)alloc((size_t)BL * DSTATE * 2);
  float*  states  = (float*) alloc((size_t)B_SZ * NCHUNK * NHEADS * HEADDIM * DSTATE * 4);
  __bf16* sinb    = (__bf16*)alloc((size_t)B_SZ * NCHUNK * NHEADS * HEADDIM * DSTATE * 2);
  float*  Ybuf    = (float*) alloc((size_t)BL * DSSM * 4);
  __bf16* ybf     = (__bf16*)alloc((size_t)BL * DSSM * 2);
  __bf16* Wout_bf = (__bf16*)alloc((size_t)DMODEL * DSSM * 2);
  float*  otmp    = (float*) alloc((size_t)BL * DMODEL * 4);

  auto blocks = [](size_t n) { return (unsigned)((n + 255) / 256); };

  // 1) casts
  k_f32_to_bf16<<<blocks((size_t)BL * DMODEL), 256, 0, stream>>>(u, u_bf, BL * DMODEL);
  k_cast_pad_rows<<<blocks((size_t)NPAD * DMODEL), 256, 0, stream>>>(W_in, Win_bf,
                                                                     DINP, DMODEL, NPAD);
  k_f32_to_bf16<<<blocks((size_t)DMODEL * DSSM), 256, 0, stream>>>(W_out, Wout_bf,
                                                                   DMODEL * DSSM);
  // 2) in-proj GEMM: zx[BL x NPAD] = u_bf[BL x DMODEL] * Win_bf^T  (TDM-fed)
  k_gemm_tdm<<<dim3(BL / 64, NPAD / 128), 256, 0, stream>>>(u_bf, Win_bf, zx,
                                                            DMODEL, NPAD);
  // 3) conv + activations
  k_conv<<<blocks((size_t)BL * CONVD), 256, 0, stream>>>(zx, conv_w, conv_b, xc);
  k_dt<<<blocks((size_t)BL * NHEADS), 256, 0, stream>>>(zx, dt_bias, A_log, dtv, dA);
  k_acum<<<blocks((size_t)B_SZ * NHEADS * NCHUNK), 256, 0, stream>>>(dA, Acum, Atot);
  k_prep<<<blocks((size_t)BL * CONVD), 256, 0, stream>>>(xc, dtv, xdt, Bbf, Cbf);
  // 4) chunk states + inter-chunk scan
  k_states<<<B_SZ * NCHUNK * NHEADS, 256, 0, stream>>>(xdt, Bbf, Acum, Atot, states);
  k_scan<<<blocks((size_t)B_SZ * NHEADS * HEADDIM * DSTATE), 256, 0, stream>>>(
      states, Atot, sinb);
  // 5) SSD core
  k_ssd<<<B_SZ * NCHUNK * NHEADS, 256, 0, stream>>>(Cbf, Bbf, xdt, sinb, Acum, xc,
                                                    Dv, Ybuf);
  // 6) gated RMSNorm -> bf16
  k_gatenorm<<<BL, 256, 0, stream>>>(Ybuf, zx, norm_w, ybf);
  // 7) out-proj GEMM: otmp[BL x DMODEL] = ybf[BL x DSSM] * Wout_bf^T  (TDM-fed)
  k_gemm_tdm<<<dim3(BL / 64, DMODEL / 128), 256, 0, stream>>>(ybf, Wout_bf, otmp,
                                                              DSSM, DMODEL);
  // 8) out = otmp + reverse_L(otmp)
  k_final<<<blocks((size_t)BL * DMODEL), 256, 0, stream>>>(otmp, out);
}
